// E3Conv2_27754078667285
// MI455X (gfx1250) — matrix operand order repaired and checked by reading, not compiled, not run on hardware
//
#include <hip/hip_runtime.h>
#include <math.h>

// ---------------------------------------------------------------------------
// Problem constants (mirror the reference)
// ---------------------------------------------------------------------------
#define N_EDGES   160000
#define N_NODES   10000
#define IN_DIM    144          // 16*(1+3+5)
#define HDIM      64
#define NPATH     11
#define WNUMEL    (NPATH*256)  // 2816
#define R_MAX     6.0f
#define NB        8

// alphas = sqrt(DIMS[l3]) / sqrt(FAN_IN[l3]);  FAN_IN = {48,64,64}
#define ALPHA0 0.14433756729740644f   // 1/sqrt(48)
#define ALPHA1 0.21650635094610965f   // sqrt(3)/8
#define ALPHA2 0.27950849718747373f   // sqrt(5)/8

typedef __attribute__((ext_vector_type(16))) _Float16 v16h;
typedef __attribute__((ext_vector_type(8)))  float    v8f;
typedef __attribute__((ext_vector_type(4)))  float    f4;

union V16 { f4 q[2]; v16h h; };

// Workspace layout (bytes)
#define WS_W3J   0            // 363 floats (padded reads guarded)
#define WS_CNT   4096         // 10000 floats
#define WS_W3T   65536        // 2816*64 halfs = 360448 B
#define WS_H2    524288       // 160000*64 halfs = 20.48 MB
#define WS_SH    22020096     // 160000*9 floats = 5.76 MB

// Path tables: PATHS enumerated exactly like the reference
// p:        0  1  2  3  4  5  6  7  8  9 10
// l1:       0  0  0  1  1  1  1  2  2  2  2
// l2:       0  1  2  0  1  1  2  0  1  2  2
// l3:       0  1  2  1  0  2  1  2  1  0  2
// w3j off:  0  1 10 35 44 53 98 143 168 213 238   (total 363)
// row off:  0  1  4  9 12 13 18  21  26  29  30   (total 35)

// ---------------------------------------------------------------------------
// Kernel 1: generate the 11 real Wigner-3j tensors on device (fp64 math)
// ---------------------------------------------------------------------------
__device__ double factd(int n) { double f = 1.0; for (int i = 2; i <= n; ++i) f *= i; return f; }

__device__ double su2cg(int j1, int m1, int j2, int m2, int j3, int m3) {
  if (m3 != m1 + m2) return 0.0;
  int vmin = -j1 + j2 + m3; if (-j1 + m1 > vmin) vmin = -j1 + m1; if (vmin < 0) vmin = 0;
  int vmax = j2 + j3 + m1; if (j3 - j1 + j2 < vmax) vmax = j3 - j1 + j2; if (j3 + m3 < vmax) vmax = j3 + m3;
  double C = sqrt((double)(2 * j3 + 1)
                  * factd(j3 + j1 - j2) * factd(j3 - j1 + j2) * factd(j1 + j2 - j3) / factd(j1 + j2 + j3 + 1)
                  * factd(j3 + m3) * factd(j3 - m3)
                  / (factd(j1 - m1) * factd(j1 + m1) * factd(j2 - m2) * factd(j2 + m2)));
  double S = 0.0;
  for (int v = vmin; v <= vmax; ++v) {
    double sgn = ((v + j2 + m2) & 1) ? -1.0 : 1.0;
    S += sgn / factd(v) * factd(j2 + j3 + m1 - v) * factd(j1 - m1 + v)
         / factd(j3 - j1 + j2 - v) / factd(j3 + m3 - v) / factd(v + j1 - j2 - m3);
  }
  return C * S;
}

__device__ void build_q(int l, double qr[5][5], double qi[5][5]) {
  for (int a = 0; a < 5; ++a) for (int b = 0; b < 5; ++b) { qr[a][b] = 0.0; qi[a][b] = 0.0; }
  const double is2 = 0.7071067811865475244;
  for (int m = -l; m < 0; ++m) { qr[l + m][l - m] = is2; qi[l + m][l + m] = -is2; }
  qr[l][l] = 1.0;
  for (int m = 1; m <= l; ++m) {
    double s = (m & 1) ? -is2 : is2;
    qr[l + m][l + m] = s; qi[l + m][l - m] = s;
  }
  // multiply by (-i)^l : l=0 -> 1, l=1 -> -i, l=2 -> -1
  if (l == 1) {
    for (int a = 0; a < 5; ++a) for (int b = 0; b < 5; ++b) {
      double r0 = qr[a][b], i0 = qi[a][b];
      qr[a][b] = i0; qi[a][b] = -r0;      // (r+ii)*(-i) -> (i0, -r0)
    }
  } else if (l == 2) {
    for (int a = 0; a < 5; ++a) for (int b = 0; b < 5; ++b) { qr[a][b] = -qr[a][b]; qi[a][b] = -qi[a][b]; }
  }
}

__global__ void k_w3j(float* __restrict__ w3jout) {
  const int p = threadIdx.x;
  if (p >= NPATH) return;
  const int PL1[NPATH] = {0,0,0,1,1,1,1,2,2,2,2};
  const int PL2[NPATH] = {0,1,2,0,1,1,2,0,1,2,2};
  const int PL3[NPATH] = {0,1,2,1,0,2,1,2,1,0,2};
  const int OFF[NPATH] = {0,1,10,35,44,53,98,143,168,213,238};
  const int l1 = PL1[p], l2 = PL2[p], l3 = PL3[p];
  const int d1 = 2 * l1 + 1, d2 = 2 * l2 + 1, d3 = 2 * l3 + 1;

  double C[5][5][5];
  for (int i = 0; i < d1; ++i)
    for (int k = 0; k < d2; ++k)
      for (int n = 0; n < d3; ++n)
        C[i][k][n] = su2cg(l1, i - l1, l2, k - l2, l3, n - l3);

  double q1r[5][5], q1i[5][5], q2r[5][5], q2i[5][5], q3r[5][5], q3i[5][5];
  build_q(l1, q1r, q1i); build_q(l2, q2r, q2i); build_q(l3, q3r, q3i);

  double Cr[5][5][5];
  double nrm2 = 0.0;
  for (int j = 0; j < d1; ++j)
    for (int l = 0; l < d2; ++l)
      for (int m = 0; m < d3; ++m) {
        double s = 0.0;
        for (int i = 0; i < d1; ++i)
          for (int k = 0; k < d2; ++k)
            for (int n = 0; n < d3; ++n) {
              double cv = C[i][k][n];
              if (cv == 0.0) continue;
              double ar = q1r[i][j], ai = q1i[i][j];
              double br = q2r[k][l], bi = q2i[k][l];
              double abr = ar * br - ai * bi;
              double abi = ar * bi + ai * br;
              // * conj(q3[n][m]) and take real part
              s += cv * (abr * q3r[n][m] + abi * q3i[n][m]);
            }
        Cr[j][l][m] = s; nrm2 += s * s;
      }
  double inv = 1.0 / sqrt(nrm2);
  for (int j = 0; j < d1; ++j)
    for (int l = 0; l < d2; ++l)
      for (int m = 0; m < d3; ++m)
        w3jout[OFF[p] + (j * d2 + l) * d3 + m] = (float)(Cr[j][l][m] * inv);
}

// ---------------------------------------------------------------------------
// Kernel 2: zero output + counters
// ---------------------------------------------------------------------------
__global__ void k_zero(float* __restrict__ out, float* __restrict__ cnt) {
  int i = blockIdx.x * 256 + threadIdx.x;
  if (i < N_NODES * IN_DIM) out[i] = 0.0f;
  if (i < N_NODES) cnt[i] = 0.0f;
}

// ---------------------------------------------------------------------------
// Kernel 3: transpose w3 (64 x 2816 f32) -> w3T[col][k] fp16
// ---------------------------------------------------------------------------
__global__ void k_w3t(const float* __restrict__ w3, _Float16* __restrict__ w3T) {
  int i = blockIdx.x * 256 + threadIdx.x;
  if (i < HDIM * WNUMEL) {
    int k = i / WNUMEL, c = i - k * WNUMEL;
    w3T[(size_t)c * HDIM + k] = (_Float16)w3[i];
  }
}

// ---------------------------------------------------------------------------
// Kernel 4: per-edge geometry + SH + radial + 3-layer silu MLP -> h (fp16)
// LDS row stride padded to 65 floats: bank = (lane + i) mod 64 -> conflict-free
// ---------------------------------------------------------------------------
#define HSTRIDE 65

__device__ __forceinline__ float silu(float x) { return x / (1.0f + expf(-x)); }

__global__ __launch_bounds__(128) void k_prep(
    const float* __restrict__ pos, const int* __restrict__ batch,
    const int* __restrict__ esrc, const int* __restrict__ edst,
    const float* __restrict__ eshift, const float* __restrict__ cellm,
    const float* __restrict__ w0, const float* __restrict__ b0,
    const float* __restrict__ w1, const float* __restrict__ b1,
    const float* __restrict__ w2, const float* __restrict__ b2,
    float* __restrict__ shw, _Float16* __restrict__ h2, float* __restrict__ cnt)
{
  __shared__ float hb[128 * HSTRIDE];   // ~33 KB, one padded 64-row per thread
  const int e = blockIdx.x * 128 + threadIdx.x;
  if (e >= N_EDGES) return;

  const int src = esrc[e], dst = edst[e];
  const int bg = batch[src];
  const float s0 = eshift[e * 3 + 0], s1 = eshift[e * 3 + 1], s2 = eshift[e * 3 + 2];
  float v[3];
#pragma unroll
  for (int j = 0; j < 3; ++j)
    v[j] = pos[dst * 3 + j] - pos[src * 3 + j]
         + s0 * cellm[bg * 9 + 0 + j] + s1 * cellm[bg * 9 + 3 + j] + s2 * cellm[bg * 9 + 6 + j];
  const float r = sqrtf(v[0] * v[0] + v[1] * v[1] + v[2] * v[2] + 1e-12f);
  const float ir = 1.0f / r;
  const float x = v[0] * ir, y = v[1] * ir, z = v[2] * ir;

  const float s3 = 1.7320508075688772f, s15 = 3.872983346207417f, s5 = 2.23606797749979f;
  float sh[9];
  sh[0] = 1.0f;
  sh[1] = s3 * y; sh[2] = s3 * z; sh[3] = s3 * x;
  sh[4] = s15 * x * y; sh[5] = s15 * y * z;
  sh[6] = 0.5f * s5 * (3.0f * z * z - 1.0f);
  sh[7] = s15 * x * z;
  sh[8] = 0.5f * s15 * (x * x - y * y);
#pragma unroll
  for (int j = 0; j < 9; ++j) shw[(size_t)e * 9 + j] = sh[j];

  // radial embedding
  float xr = r * (1.0f / R_MAX); if (xr > 1.0f) xr = 1.0f;
  const float gate = (r <= R_MAX) ? 2.8284271247461903f : 0.0f; // sqrt(8)
  float emb[NB];
#pragma unroll
  for (int k = 0; k < NB; ++k) {
    float d = (xr - (float)k * (1.0f / 7.0f)) * 7.0f;
    emb[k] = expf(-0.5f * d * d) * gate;
  }

  float g[HDIM];
  float* hrow = hb + threadIdx.x * HSTRIDE;
  // layer 0: 8 -> 64
  for (int o = 0; o < HDIM; ++o) {
    float a = b0[o];
#pragma unroll
    for (int i = 0; i < NB; ++i) a += emb[i] * w0[i * HDIM + o];
    g[o] = silu(a);
  }
  for (int o = 0; o < HDIM; ++o) hrow[o] = g[o];
  // layer 1: 64 -> 64
  for (int o = 0; o < HDIM; ++o) {
    float a = b1[o];
#pragma unroll 8
    for (int i = 0; i < HDIM; ++i) a += hrow[i] * w1[i * HDIM + o];
    g[o] = silu(a);
  }
  for (int o = 0; o < HDIM; ++o) hrow[o] = g[o];
  // layer 2: 64 -> 64
  for (int o = 0; o < HDIM; ++o) {
    float a = b2[o];
#pragma unroll 8
    for (int i = 0; i < HDIM; ++i) a += hrow[i] * w2[i * HDIM + o];
    g[o] = silu(a);
  }
  for (int o = 0; o < HDIM; ++o) h2[(size_t)e * HDIM + o] = (_Float16)g[o];

  atomicAdd(cnt + dst, 1.0f);
}

// ---------------------------------------------------------------------------
// Kernel 5: fused  w = h@w3+b3 (WMMA)  ->  tensor product  ->  atomic scatter
// One wave (32 threads) per 16 edges.
// ---------------------------------------------------------------------------
template<int L1, int L2, int L3>
__device__ __forceinline__ void computeY(const float* __restrict__ x1t,
                                         const float* shl,
                                         float* __restrict__ yT,
                                         const float* __restrict__ w3jl,
                                         int w3joff, int rp, float alpha,
                                         int e, int uh)
{
  constexpr int D1 = 2 * L1 + 1, D2 = 2 * L2 + 1, D3 = 2 * L3 + 1;
  constexpr int XOFF = (L1 == 0) ? 0 : ((L1 == 1) ? 16 : 64);
  constexpr int SOFF = (L2 == 0) ? 0 : ((L2 == 1) ? 1 : 4);
  float acc[D3][8];
#pragma unroll
  for (int k = 0; k < D3; ++k)
#pragma unroll
    for (int u = 0; u < 8; ++u) acc[k][u] = 0.0f;
#pragma unroll
  for (int i = 0; i < D1; ++i) {
    float xv[8];
#pragma unroll
    for (int u = 0; u < 8; ++u) xv[u] = x1t[e * IN_DIM + XOFF + i * 16 + uh + u];
#pragma unroll
    for (int j = 0; j < D2; ++j) {
      float s = shl[SOFF + j];
      float xs[8];
#pragma unroll
      for (int u = 0; u < 8; ++u) xs[u] = xv[u] * s;
#pragma unroll
      for (int k = 0; k < D3; ++k) {
        float w = w3jl[w3joff + (i * D2 + j) * D3 + k];
#pragma unroll
        for (int u = 0; u < 8; ++u) acc[k][u] += xs[u] * w;
      }
    }
  }
#pragma unroll
  for (int k = 0; k < D3; ++k)
#pragma unroll
    for (int u = 0; u < 8; ++u)
      yT[(rp + k) * 256 + (uh + u) * 16 + e] = alpha * acc[k][u];
}

template<int L3>
__device__ __forceinline__ void pathWmma(int p, int rp,
    const _Float16* __restrict__ w3T, const float* __restrict__ b3,
    const float* __restrict__ yT, v16h aLo, v16h aHi, float* accO)
{
  constexpr int D3 = 2 * L3 + 1;
  constexpr int K0 = (L3 == 0) ? 0 : ((L3 == 1) ? 1 : 4);
  const int lane = threadIdx.x;
  const int wcol = lane & 15;
  const int kb = (lane & 16) >> 1;   // 0 or 8: B-operand K-offset within chunk
  const int eh = (lane & 16) >> 1;   // 0 or 8 : which 8 output rows this lane holds
  // Moderate unroll: keep ~8 B-loads in flight without exploding VGPR usage
#pragma unroll 4
  for (int t = 0; t < 16; ++t) {
    const int col = p * 256 + t * 16 + wcol;      // flattened (u=t, w=wcol)
    V16 b0u, b1u;
    const f4* bp0 = (const f4*)(w3T + (size_t)col * HDIM + kb);
    b0u.q[0] = bp0[0]; b0u.q[1] = bp0[2];         // K = kb..kb+7, 16+kb..16+kb+7  (chunk 0)
    const f4* bp1 = (const f4*)(w3T + (size_t)col * HDIM + 32 + kb);
    b1u.q[0] = bp1[0]; b1u.q[1] = bp1[2];         // K = 32+kb.., 48+kb..   (chunk 1)
    const float bias = b3[col];
    v8f c;
#pragma unroll
    for (int q = 0; q < 8; ++q) c[q] = bias;
    c = __builtin_amdgcn_wmma_f32_16x16x32_f16(false, aLo, false, b0u.h, (short)0, c, false, false);
    c = __builtin_amdgcn_wmma_f32_16x16x32_f16(false, aHi, false, b1u.h, (short)0, c, false, false);
    // wp[e = eh+r, u = t, w = wcol] = c[r]  ->  TP accumulation
#pragma unroll
    for (int r = 0; r < 8; ++r) {
#pragma unroll
      for (int k = 0; k < D3; ++k)
        accO[r * 9 + K0 + k] += yT[(rp + k) * 256 + t * 16 + (eh + r)] * c[r];
    }
  }
}

__global__ __launch_bounds__(32) void k_tp(
    const float* __restrict__ f_in,
    const int* __restrict__ esrc, const int* __restrict__ edst,
    const _Float16* __restrict__ h2, const float* __restrict__ shw,
    const _Float16* __restrict__ w3T, const float* __restrict__ b3,
    const float* __restrict__ w3j, float* __restrict__ out)
{
  __shared__ float yT[35 * 256];      // y[row][u][e], alpha folded in
  __shared__ float x1t[16 * IN_DIM];  // gathered source features
  __shared__ float w3jl[384];
  __shared__ int   dstIds[16];

  const int tid = threadIdx.x;
  const int e0 = blockIdx.x * 16;

  for (int i = tid; i < 384; i += 32) w3jl[i] = (i < 363) ? w3j[i] : 0.0f;
  if (tid < 16) dstIds[tid] = edst[e0 + tid];
  for (int idx = tid; idx < 16 * IN_DIM; idx += 32) {
    int e = idx / IN_DIM, c = idx - e * IN_DIM;
    x1t[idx] = f_in[(size_t)esrc[e0 + e] * IN_DIM + c];
  }
  __syncthreads();

  const int e  = tid & 15;
  const int uh = (tid & 16) >> 1;   // 0 or 8: which 8 mul-channels this lane builds
  float shl[9];
#pragma unroll
  for (int j = 0; j < 9; ++j) shl[j] = shw[(size_t)(e0 + e) * 9 + j];

  // y[e,k,u] for all 11 paths (alpha folded)
  computeY<0,0,0>(x1t, shl, yT, w3jl,   0,  0, ALPHA0, e, uh);
  computeY<0,1,1>(x1t, shl, yT, w3jl,   1,  1, ALPHA1, e, uh);
  computeY<0,2,2>(x1t, shl, yT, w3jl,  10,  4, ALPHA2, e, uh);
  computeY<1,0,1>(x1t, shl, yT, w3jl,  35,  9, ALPHA1, e, uh);
  computeY<1,1,0>(x1t, shl, yT, w3jl,  44, 12, ALPHA0, e, uh);
  computeY<1,1,2>(x1t, shl, yT, w3jl,  53, 13, ALPHA2, e, uh);
  computeY<1,2,1>(x1t, shl, yT, w3jl,  98, 18, ALPHA1, e, uh);
  computeY<2,0,2>(x1t, shl, yT, w3jl, 143, 21, ALPHA2, e, uh);
  computeY<2,1,1>(x1t, shl, yT, w3jl, 168, 26, ALPHA1, e, uh);
  computeY<2,2,0>(x1t, shl, yT, w3jl, 213, 29, ALPHA0, e, uh);
  computeY<2,2,2>(x1t, shl, yT, w3jl, 238, 30, ALPHA2, e, uh);
  __syncthreads();

  // A operand (h tile, fp16): lane holds row M=e; halfs per ISA 16-bit A layout
  const _Float16* hp = h2 + (size_t)(e0 + e) * HDIM + ((tid & 16) ? 8 : 0);
  V16 a0u, a1u;
  a0u.q[0] = *(const f4*)(hp +  0);  a0u.q[1] = *(const f4*)(hp + 16);  // K chunk [0,32)
  a1u.q[0] = *(const f4*)(hp + 32);  a1u.q[1] = *(const f4*)(hp + 48);  // K chunk [32,64)

  float accO[8 * 9];
#pragma unroll
  for (int i = 0; i < 72; ++i) accO[i] = 0.0f;

  pathWmma<0>( 0,  0, w3T, b3, yT, a0u.h, a1u.h, accO);
  pathWmma<1>( 1,  1, w3T, b3, yT, a0u.h, a1u.h, accO);
  pathWmma<2>( 2,  4, w3T, b3, yT, a0u.h, a1u.h, accO);
  pathWmma<1>( 3,  9, w3T, b3, yT, a0u.h, a1u.h, accO);
  pathWmma<0>( 4, 12, w3T, b3, yT, a0u.h, a1u.h, accO);
  pathWmma<2>( 5, 13, w3T, b3, yT, a0u.h, a1u.h, accO);
  pathWmma<1>( 6, 18, w3T, b3, yT, a0u.h, a1u.h, accO);
  pathWmma<2>( 7, 21, w3T, b3, yT, a0u.h, a1u.h, accO);
  pathWmma<1>( 8, 26, w3T, b3, yT, a0u.h, a1u.h, accO);
  pathWmma<0>( 9, 29, w3T, b3, yT, a0u.h, a1u.h, accO);
  pathWmma<2>(10, 30, w3T, b3, yT, a0u.h, a1u.h, accO);

  // scatter: lane owns edges e = eh..eh+7, column w = tid&15, all 9 k-rows
  const int wcol = tid & 15;
  const int eh = (tid & 16) >> 1;
#pragma unroll
  for (int r = 0; r < 8; ++r) {
    const int node = dstIds[eh + r];
    float* po = out + (size_t)node * IN_DIM + wcol;
#pragma unroll
    for (int k = 0; k < 9; ++k) atomicAdd(po + k * 16, accO[r * 9 + k]);
  }
}

// ---------------------------------------------------------------------------
// Kernel 6: divide by clamped in-degree
// ---------------------------------------------------------------------------
__global__ void k_div(float* __restrict__ out, const float* __restrict__ cnt) {
  int i = blockIdx.x * 256 + threadIdx.x;
  if (i < N_NODES * IN_DIM) {
    float c = cnt[i / IN_DIM];
    out[i] = out[i] / fmaxf(c, 1.0f);
  }
}

// ---------------------------------------------------------------------------
extern "C" void kernel_launch(void* const* d_in, const int* in_sizes, int n_in,
                              void* d_out, int out_size, void* d_ws, size_t ws_size,
                              hipStream_t stream) {
  (void)in_sizes; (void)n_in; (void)out_size; (void)ws_size;
  const float* f_in   = (const float*)d_in[0];
  const float* pos    = (const float*)d_in[1];
  /* d_in[2] = A (unused by reference) */
  const int*   batch  = (const int*)d_in[3];
  const int*   esrc   = (const int*)d_in[4];
  const int*   edst   = (const int*)d_in[5];
  const float* eshift = (const float*)d_in[6];
  const float* cellm  = (const float*)d_in[7];
  const float* w0 = (const float*)d_in[8];
  const float* b0 = (const float*)d_in[9];
  const float* w1 = (const float*)d_in[10];
  const float* b1 = (const float*)d_in[11];
  const float* w2 = (const float*)d_in[12];
  const float* b2 = (const float*)d_in[13];
  const float* w3 = (const float*)d_in[14];
  const float* b3 = (const float*)d_in[15];
  float* out = (float*)d_out;

  char* ws = (char*)d_ws;
  float*     w3j = (float*)(ws + WS_W3J);
  float*     cnt = (float*)(ws + WS_CNT);
  _Float16*  w3T = (_Float16*)(ws + WS_W3T);
  _Float16*  h2  = (_Float16*)(ws + WS_H2);
  float*     shw = (float*)(ws + WS_SH);

  k_w3j<<<1, 32, 0, stream>>>(w3j);
  k_zero<<<(N_NODES * IN_DIM + 255) / 256, 256, 0, stream>>>(out, cnt);
  k_w3t<<<(HDIM * WNUMEL + 255) / 256, 256, 0, stream>>>(w3, w3T);
  k_prep<<<(N_EDGES + 127) / 128, 128, 0, stream>>>(pos, batch, esrc, edst, eshift, cellm,
                                                    w0, b0, w1, b1, w2, b2, shw, h2, cnt);
  k_tp<<<N_EDGES / 16, 32, 0, stream>>>(f_in, esrc, edst, h2, shw, w3T, b3, w3j, out);
  k_div<<<(N_NODES * IN_DIM + 255) / 256, 256, 0, stream>>>(out, cnt);
}
